// EncoderDecoder_75230647156740
// MI455X (gfx1250) — matrix operand order repaired and checked by reading, not compiled, not run on hardware
//
#include <hip/hip_runtime.h>
#include <hip/hip_bf16.h>

// ---------------- problem constants ----------------
#define L_SEQ   512
#define D_IN    1024
#define D_H     2048
#define D_G     8192      // 4*H (gate vector length)
#define N_VOCAB 50257
#define LSTM_BLOCKS 64

typedef __attribute__((ext_vector_type(16))) __bf16 v16bf;
typedef __attribute__((ext_vector_type(8)))  float  v8f;
typedef __attribute__((ext_vector_type(8)))  __bf16 bf16x8;

__device__ __forceinline__ float sigmoidf_(float x) { return 1.0f / (1.0f + __expf(-x)); }

__device__ __forceinline__ v16bf cvt16(float4 a, float4 b, float4 c, float4 d) {
  v16bf r;
  r[0]=(__bf16)a.x;  r[1]=(__bf16)a.y;  r[2]=(__bf16)a.z;  r[3]=(__bf16)a.w;
  r[4]=(__bf16)b.x;  r[5]=(__bf16)b.y;  r[6]=(__bf16)b.z;  r[7]=(__bf16)b.w;
  r[8]=(__bf16)c.x;  r[9]=(__bf16)c.y;  r[10]=(__bf16)c.z; r[11]=(__bf16)c.w;
  r[12]=(__bf16)d.x; r[13]=(__bf16)d.y; r[14]=(__bf16)d.z; r[15]=(__bf16)d.w;
  return r;
}

__device__ __forceinline__ v16bf zero16() {
  v16bf r;
  #pragma unroll
  for (int i = 0; i < 16; ++i) r[i] = (__bf16)0.0f;
  return r;
}

// ---------------- f32 -> bf16 weight conversion (W_hh stays hot in L2 at half size) ----
__global__ void seq2seq_cvt_bf16(const float* __restrict__ src, __bf16* __restrict__ dst, int n) {
  int i = blockIdx.x * blockDim.x + threadIdx.x;
  int stride = gridDim.x * blockDim.x;
  for (; i < n; i += stride) dst[i] = (__bf16)src[i];
}

// ---------------- init: zero h state + barrier words -------------------------
__global__ void seq2seq_init(float* __restrict__ h, unsigned* __restrict__ bar) {
  int i = blockIdx.x * blockDim.x + threadIdx.x;
  if (i < D_H) h[i] = 0.0f;
  if (i == 0) { bar[0] = 0u; bar[1] = 0u; }
}

// ---------------- bf16 WMMA GEMM:  C[512][N] = A[512][K] * Bt[N][K]^T + bias0 + bias1 ---
// grid.x = ceil(N/64); block = 256 (8 waves). Wave w owns rows [w*64, w*64+64).
__global__ __launch_bounds__(256) void seq2seq_gemm_wmma(
    const float* __restrict__ A,     // [512][K]  row-major
    const float* __restrict__ Bt,    // [N][K]    row n contiguous in k
    const float* __restrict__ bias0, // [N] or null
    const float* __restrict__ bias1, // [N] or null
    float* __restrict__ C,           // [512][N]
    int N, int K)
{
  const int lane = threadIdx.x & 31;
  const int wave = threadIdx.x >> 5;
  const int l15  = lane & 15;
  const int hi   = lane >> 4;          // 0: lanes 0-15, 1: lanes 16-31
  const int Nblk = blockIdx.x * 64;
  const int M0   = wave * 64;

  v8f acc[4][4];
  #pragma unroll
  for (int mt = 0; mt < 4; ++mt)
    #pragma unroll
    for (int nt = 0; nt < 4; ++nt)
      #pragma unroll
      for (int j = 0; j < 8; ++j) acc[mt][nt][j] = 0.0f;

  for (int kk = 0; kk < K; kk += 32) {
    // ---- B fragments: element (k,n); lane holds col n=l15, K range hi?16-31:0-15 ----
    v16bf bfrag[4];
    #pragma unroll
    for (int nt = 0; nt < 4; ++nt) {
      int n = Nblk + nt * 16 + l15;
      if (n < N) {
        const float4* p = (const float4*)(Bt + (size_t)n * K + kk + hi * 16);
        __builtin_prefetch(p + 8, 0, 1);   // next k-chunk -> global_prefetch_b8
        bfrag[nt] = cvt16(p[0], p[1], p[2], p[3]);
      } else {
        bfrag[nt] = zero16();
      }
    }
    // ---- A fragments + 16 WMMAs ----
    #pragma unroll
    for (int mt = 0; mt < 4; ++mt) {
      int m = M0 + mt * 16 + l15;
      const float* ap = A + (size_t)m * K + kk + hi * 16;  // K 0-7 (or 16-23)
      const float4* p0 = (const float4*)(ap);
      const float4* p1 = (const float4*)(ap + 8);           // K 8-15 (or 24-31)
      v16bf afrag = cvt16(p0[0], p0[1], p1[0], p1[1]);
      #pragma unroll
      for (int nt = 0; nt < 4; ++nt)
        acc[mt][nt] = __builtin_amdgcn_wmma_f32_16x16x32_bf16(
            false, afrag, false, bfrag[nt], (short)0, acc[mt][nt], false, false);
    }
  }

  // ---- epilogue: C/D layout: lane n=l15, m = M0t + hi*8 + vgpr ----
  #pragma unroll
  for (int nt = 0; nt < 4; ++nt) {
    int n = Nblk + nt * 16 + l15;
    if (n >= N) continue;
    float bs = (bias0 ? bias0[n] : 0.0f) + (bias1 ? bias1[n] : 0.0f);
    #pragma unroll
    for (int mt = 0; mt < 4; ++mt)
      #pragma unroll
      for (int j = 0; j < 8; ++j) {
        int m = M0 + mt * 16 + hi * 8 + j;
        C[(size_t)m * N + n] = acc[mt][nt][j] + bs;
      }
  }
}

// ---------------- persistent LSTM recurrence (enc 512 steps + dec 512 steps) ----------
// 64 blocks x 256 threads. Block b owns k in [b*32, b*32+32). Wave w owns k-local w*4..w*4+3,
// computing the 4 gate rows for each of its 4 k's (16 dot products of length 2048 vs h).
__global__ __launch_bounds__(256) void seq2seq_lstm(
    const float*  __restrict__ gx_e,   // [512][8192] = src@W_ih_e^T + b_ih_e + b_hh_e
    const float*  __restrict__ gx_d,   // [512][8192]
    const __bf16* __restrict__ whh_e,  // [8192][2048] bf16
    const __bf16* __restrict__ whh_d,  // [8192][2048] bf16
    float* __restrict__ h,             // [2048] shared state
    float* __restrict__ hdec,          // [512][2048] decoder hidden trace
    unsigned* __restrict__ bar)        // [2]: monotonic counter, generation
{
  __shared__ float red[128];           // 8 waves x 16 reduced gate values
  const int lane = threadIdx.x & 31;
  const int wave = threadIdx.x >> 5;
  const int blk  = blockIdx.x;

  float c_reg = 0.0f;                  // cell state: k = blk*32 + tid is owned by one thread forever
  unsigned phase = 0;

  for (int step = 0; step < 2 * L_SEQ; ++step) {
    const bool dec = (step >= L_SEQ);
    const int  t   = dec ? (step - L_SEQ) : step;
    const float*  gx  = (dec ? gx_d : gx_e) + (size_t)t * D_G;
    const __bf16* whh = dec ? whh_d : whh_e;

    // preload this lane's slice of h: hreg[j*8+i] = h[j*256 + lane*8 + i]  (agent scope: cross-WGP)
    float hreg[64];
    #pragma unroll
    for (int j = 0; j < 8; ++j)
      #pragma unroll
      for (int i = 0; i < 8; ++i)
        hreg[j * 8 + i] = __hip_atomic_load(&h[j * 256 + lane * 8 + i],
                                            __ATOMIC_RELAXED, __HIP_MEMORY_SCOPE_AGENT);

    // 16 rows: r = gate*4 + kLocal
    #pragma unroll 1
    for (int r = 0; r < 16; ++r) {
      const int gate = r >> 2, kl = r & 3;
      const int row  = gate * D_H + blk * 32 + wave * 4 + kl;
      const bf16x8* wp = (const bf16x8*)(whh + (size_t)row * D_H);
      float sum = 0.0f;
      #pragma unroll
      for (int j = 0; j < 8; ++j) {
        bf16x8 wv = wp[j * 32 + lane];          // 16B coalesced, L2-resident
        #pragma unroll
        for (int i = 0; i < 8; ++i)
          sum = fmaf((float)wv[i], hreg[j * 8 + i], sum);
      }
      #pragma unroll
      for (int off = 16; off > 0; off >>= 1)
        sum += __shfl_xor(sum, off, 32);
      if (lane == 0) red[wave * 16 + r] = sum + gx[row];
    }
    __syncthreads();

    // cell update: threads 0..31 each own k = blk*32 + tid
    if (threadIdx.x < 32) {
      const int w2 = threadIdx.x >> 2, kl = threadIdx.x & 3;
      const int k  = blk * 32 + threadIdx.x;
      const float ig = sigmoidf_(red[w2 * 16 + 0 * 4 + kl]);
      const float fg = sigmoidf_(red[w2 * 16 + 1 * 4 + kl]);
      const float gg = tanhf    (red[w2 * 16 + 2 * 4 + kl]);
      const float og = sigmoidf_(red[w2 * 16 + 3 * 4 + kl]);
      c_reg = fg * c_reg + ig * gg;
      const float hn = og * tanhf(c_reg);
      __hip_atomic_store(&h[k], hn, __ATOMIC_RELAXED, __HIP_MEMORY_SCOPE_AGENT);
      if (dec) hdec[(size_t)t * D_H + k] = hn;
    }

    // device-wide barrier (monotonic counter + generation word)
    __syncthreads();
    if (threadIdx.x == 0) {
      unsigned arrive = __hip_atomic_fetch_add(&bar[0], 1u, __ATOMIC_ACQ_REL,
                                               __HIP_MEMORY_SCOPE_AGENT) + 1u;
      if (arrive == (phase + 1u) * (unsigned)LSTM_BLOCKS) {
        __hip_atomic_store(&bar[1], phase + 1u, __ATOMIC_RELEASE, __HIP_MEMORY_SCOPE_AGENT);
      } else {
        while (__hip_atomic_load(&bar[1], __ATOMIC_ACQUIRE, __HIP_MEMORY_SCOPE_AGENT) <= phase)
          __builtin_amdgcn_s_sleep(1);
      }
    }
    __syncthreads();
    ++phase;
  }
}

// ---------------- host side ----------------
extern "C" void kernel_launch(void* const* d_in, const int* in_sizes, int n_in,
                              void* d_out, int out_size, void* d_ws, size_t ws_size,
                              hipStream_t stream) {
  (void)in_sizes; (void)n_in; (void)out_size; (void)ws_size;
  const float* src    = (const float*)d_in[0];
  const float* tgt    = (const float*)d_in[1];
  const float* W_ih_e = (const float*)d_in[2];
  const float* W_hh_e = (const float*)d_in[3];
  const float* b_ih_e = (const float*)d_in[4];
  const float* b_hh_e = (const float*)d_in[5];
  const float* W_ih_d = (const float*)d_in[6];
  const float* W_hh_d = (const float*)d_in[7];
  const float* b_ih_d = (const float*)d_in[8];
  const float* b_hh_d = (const float*)d_in[9];
  const float* W_out  = (const float*)d_in[10];
  const float* b_out  = (const float*)d_in[11];
  float* out = (float*)d_out;

  // workspace layout
  char* p = (char*)d_ws;
  float*  gx_e     = (float*)p;  p += (size_t)L_SEQ * D_G * sizeof(float);   // 16 MB
  float*  gx_d     = (float*)p;  p += (size_t)L_SEQ * D_G * sizeof(float);   // 16 MB
  __bf16* whh_e_bf = (__bf16*)p; p += (size_t)D_G * D_H * sizeof(__bf16);    // 32 MB
  __bf16* whh_d_bf = (__bf16*)p; p += (size_t)D_G * D_H * sizeof(__bf16);    // 32 MB
  float*  hdec     = (float*)p;  p += (size_t)L_SEQ * D_H * sizeof(float);   // 4 MB
  float*  hbuf     = (float*)p;  p += (size_t)D_H * sizeof(float);
  unsigned* bar    = (unsigned*)p;

  // 1) W_hh -> bf16 (L2-resident at half the bytes for the 1024-step recurrence)
  seq2seq_cvt_bf16<<<1024, 256, 0, stream>>>(W_hh_e, whh_e_bf, D_G * D_H);
  seq2seq_cvt_bf16<<<1024, 256, 0, stream>>>(W_hh_d, whh_d_bf, D_G * D_H);

  // 2) zero h state + barrier words
  seq2seq_init<<<8, 256, 0, stream>>>(hbuf, bar);

  // 3) batched input-side GEMMs: Gx = X @ W_ih^T + b_ih + b_hh   ([512,1024]x[1024,8192])
  seq2seq_gemm_wmma<<<dim3(D_G / 64), 256, 0, stream>>>(src, W_ih_e, b_ih_e, b_hh_e, gx_e, D_G, D_IN);
  seq2seq_gemm_wmma<<<dim3(D_G / 64), 256, 0, stream>>>(tgt, W_ih_d, b_ih_d, b_hh_d, gx_d, D_G, D_IN);

  // 4) persistent encoder+decoder recurrence (grid-synced, W_hh streamed from L2 as bf16)
  seq2seq_lstm<<<LSTM_BLOCKS, 256, 0, stream>>>(gx_e, gx_d, whh_e_bf, whh_d_bf, hbuf, hdec, bar);

  // 5) vocab projection: out = H_dec @ W_out^T + b_out   ([512,2048]x[2048,50257])
  seq2seq_gemm_wmma<<<dim3((N_VOCAB + 63) / 64), 256, 0, stream>>>(hdec, W_out, b_out, nullptr,
                                                                   out, N_VOCAB, D_H);
}
